// NLblock2d_86509231276386
// MI455X (gfx1250) — matrix operand order repaired and checked by reading, not compiled, not run on hardware
//
#include <hip/hip_runtime.h>

// Problem constants (from reference): x[8,64,56,56]
#define BB 8
#define CC 64
#define NN 3136   // 56*56

typedef __attribute__((ext_vector_type(16))) __bf16 bf16x16;
typedef __attribute__((ext_vector_type(8)))  __bf16 bf16x8;
typedef __attribute__((ext_vector_type(8)))  float  f32x8;
typedef unsigned short u16;
typedef unsigned int   u32;

__device__ __forceinline__ u16 f2bf(float f) {
  union { float f; u32 u; } v; v.f = f;
  u32 r = v.u + 0x7FFFu + ((v.u >> 16) & 1u);   // round-to-nearest-even
  return (u16)(r >> 16);
}

__device__ __forceinline__ bf16x16 concat8(bf16x8 a, bf16x8 b) {
  return __builtin_shufflevector(a, b, 0,1,2,3,4,5,6,7,8,9,10,11,12,13,14,15);
}

__device__ __forceinline__ f32x8 wmma_bf16(bf16x16 a, bf16x16 b, f32x8 c) {
  // D = A(16x32 bf16) * B(32x16 bf16) + C(16x16 f32)
  return __builtin_amdgcn_wmma_f32_16x16x32_bf16(false, a, false, b, (short)0, c,
                                                 false, false);
}

// ---------------------------------------------------------------------------
// Kernel 1: QKV 1x1-conv projections, fp32 compute, bf16 outputs.
//   Qb[b][n][o] = sum_c w2[o][c] x[b][c][n]   (query, row-major over n)
//   Kb[b][n][o] = sum_c w3[o][c] x[b][c][n]   (key^T, row-major over n)
//   Vb[b][o][n] = sum_c w1[o][c] x[b][c][n]   (value, channel-major)
// ---------------------------------------------------------------------------
__global__ __launch_bounds__(256) void proj_qkv(
    const float* __restrict__ x,
    const float* __restrict__ w1, const float* __restrict__ w2,
    const float* __restrict__ w3,
    u16* __restrict__ Qb, u16* __restrict__ Kb, u16* __restrict__ Vb) {
  __shared__ float s_w[3][CC][CC];  // [proj][c][o]: transposed -> conflict-free
  __shared__ float s_x[CC][32];     // [c][j]: broadcast reads

  const int b  = blockIdx.x / (NN / 32);
  const int n0 = (blockIdx.x % (NN / 32)) * 32;
  const int t  = threadIdx.x;

  for (int i = t; i < 3 * CC * CC; i += 256) {
    const int p = i >> 12, rem = i & 4095, o = rem >> 6, c = rem & 63;
    const float* wsrc = (p == 0) ? w1 : ((p == 1) ? w2 : w3);
    s_w[p][c][o] = wsrc[rem];
  }
  for (int i = t; i < CC * 32; i += 256) {
    const int c = i >> 5, j = i & 31;
    s_x[c][j] = x[((size_t)b * CC + c) * NN + n0 + j];
  }
  __syncthreads();

  const int o = t & 63, g = t >> 6;
  u16 vbuf[8];
  for (int jj = 0; jj < 8; ++jj) {
    const int j = g * 8 + jj;
    float a1 = 0.f, a2 = 0.f, a3 = 0.f;
    #pragma unroll
    for (int c = 0; c < CC; ++c) {
      const float xv = s_x[c][j];
      a1 = fmaf(s_w[0][c][o], xv, a1);
      a2 = fmaf(s_w[1][c][o], xv, a2);
      a3 = fmaf(s_w[2][c][o], xv, a3);
    }
    const size_t nIdx = (size_t)b * NN + n0 + j;
    vbuf[jj] = f2bf(a1);
    Qb[nIdx * CC + o] = f2bf(a2);   // lanes: consecutive o -> coalesced
    Kb[nIdx * CC + o] = f2bf(a3);
  }
  // V: one 16B store of 8 contiguous tokens for this channel row
  u16* vp = Vb + ((size_t)b * CC + o) * NN + n0 + g * 8;
  uint4 pk;
  pk.x = (u32)vbuf[0] | ((u32)vbuf[1] << 16);
  pk.y = (u32)vbuf[2] | ((u32)vbuf[3] << 16);
  pk.z = (u32)vbuf[4] | ((u32)vbuf[5] << 16);
  pk.w = (u32)vbuf[6] | ((u32)vbuf[7] << 16);
  *(uint4*)vp = pk;
}

// ---------------------------------------------------------------------------
// Kernel 2: fused flash-attention. One wave32 per 16 query rows.
//   Streams keys/values in blocks of 32 tokens. Scores tile (16x32) via
//   v_wmma_f32_16x16x32_bf16, online softmax staged in LDS, ctx accumulated
//   in 4 f32 WMMA accumulators (16 rows x 64 channels).
//   V-fragment loads are issued at the top of the loop so their global
//   latency is covered by the score WMMAs + softmax phase.
// ---------------------------------------------------------------------------
__global__ __launch_bounds__(32) void attn_fused(
    const u16* __restrict__ Qb, const u16* __restrict__ Kb,
    const u16* __restrict__ Vb, float* __restrict__ ctx) {
  const int b  = blockIdx.x / (NN / 16);
  const int rt = blockIdx.x % (NN / 16);
  const int n0 = rt * 16;
  const int L  = threadIdx.x;
  const int r  = L & 15;    // row in A tiles, col in B/C tiles
  const int hi = L >> 4;    // lane-half selector

  __shared__ float s_S[16][33];      // score tile staging (+1 pad)
  __shared__ u16   s_P[16][32];      // exp(S - m) as bf16, row-major
  __shared__ float s_m[16], s_l[16], s_alpha[16];
  __shared__ float s_red[2][16];

  if (L < 16) { s_m[L] = -__builtin_inff(); s_l[L] = 0.f; }

  // Q A-fragments: lane holds contiguous K-run [16hi,16hi+16)
  const u16* qrow = Qb + ((size_t)b * NN + n0 + r) * CC;
  bf16x16 aQ0, aQ1;
  {
    bf16x8 l0 = *(const bf16x8*)(qrow + 16 * hi);
    bf16x8 h0 = *(const bf16x8*)(qrow + 8 + 16 * hi);
    aQ0 = concat8(l0, h0);
    bf16x8 l1 = *(const bf16x8*)(qrow + 32 + 16 * hi);
    bf16x8 h1 = *(const bf16x8*)(qrow + 40 + 16 * hi);
    aQ1 = concat8(l1, h1);
  }

  f32x8 acc0 = {0.f,0.f,0.f,0.f,0.f,0.f,0.f,0.f};
  f32x8 acc1 = acc0, acc2 = acc0, acc3 = acc0;

  const u16* kbase = Kb + (size_t)b * NN * CC;
  const u16* vbase = Vb + (size_t)b * CC * NN + (size_t)r * NN + 16 * hi;
  __syncthreads();

  for (int c0 = 0; c0 < NN; c0 += 32) {
    // ---- V B-fragments first: latency hidden behind scores + softmax ------
    const u16* vp = vbase + c0;
    bf16x16 bv0 = *(const bf16x16*)(vp);
    bf16x16 bv1 = *(const bf16x16*)(vp + (size_t)16 * NN);
    bf16x16 bv2 = *(const bf16x16*)(vp + (size_t)32 * NN);
    bf16x16 bv3 = *(const bf16x16*)(vp + (size_t)48 * NN);

    // Prefetch next block's K/V (speculative; dropped if beyond mapping,
    // address stays inside the workspace for all iterations).
    __builtin_prefetch(kbase + (size_t)(c0 + 32 + r) * CC, 0, 3);
    __builtin_prefetch(vbase + c0 + 32, 0, 3);

    // ---- scores: S(16x32) = Q(16x64) * K^T(64x32), two 16-col subtiles ----
    f32x8 s0 = {0.f,0.f,0.f,0.f,0.f,0.f,0.f,0.f};
    f32x8 s1 = s0;
    {
      const u16* kp0 = kbase + (size_t)(c0 + r) * CC + 16 * hi;
      const u16* kp1 = kbase + (size_t)(c0 + 16 + r) * CC + 16 * hi;
      bf16x16 b00 = *(const bf16x16*)(kp0);        // c in [16hi, 16hi+16)
      bf16x16 b01 = *(const bf16x16*)(kp0 + 32);   // c in [32+16hi, ...)
      bf16x16 b10 = *(const bf16x16*)(kp1);
      bf16x16 b11 = *(const bf16x16*)(kp1 + 32);
      s0 = wmma_bf16(aQ0, b00, s0);
      s0 = wmma_bf16(aQ1, b01, s0);
      s1 = wmma_bf16(aQ0, b10, s1);
      s1 = wmma_bf16(aQ1, b11, s1);
    }
    // stage S: D layout row = v + 8*hi, col = r
    #pragma unroll
    for (int v = 0; v < 8; ++v) {
      s_S[v + 8 * hi][r]      = s0[v];
      s_S[v + 8 * hi][16 + r] = s1[v];
    }
    __syncthreads();

    // partial row max (lane -> row r, cols [16hi, 16hi+16))
    float mx = -__builtin_inff();
    #pragma unroll
    for (int i = 0; i < 16; ++i) mx = fmaxf(mx, s_S[r][16 * hi + i]);
    s_red[hi][r] = mx;
    __syncthreads();

    if (L < 16) {
      const float mo = s_m[L];
      const float mn = fmaxf(mo, fmaxf(s_red[0][L], s_red[1][L]));
      s_alpha[L] = __expf(mo - mn);   // exp(-inf)=0 on first block
      s_m[L] = mn;
    }
    __syncthreads();

    // exp, bf16 P tile, partial row sums
    {
      const float mn = s_m[r];
      float ls = 0.f;
      #pragma unroll
      for (int i = 0; i < 16; ++i) {
        const float p = __expf(s_S[r][16 * hi + i] - mn);
        ls += p;
        s_P[r][16 * hi + i] = f2bf(p);
      }
      s_red[hi][r] = ls;
    }
    __syncthreads();

    if (L < 16) s_l[L] = s_l[L] * s_alpha[L] + s_red[0][L] + s_red[1][L];

    // rescale ctx accumulators by per-row alpha
    float al[8];
    #pragma unroll
    for (int v = 0; v < 8; ++v) al[v] = s_alpha[v + 8 * hi];
    #pragma unroll
    for (int v = 0; v < 8; ++v) {
      acc0[v] *= al[v]; acc1[v] *= al[v]; acc2[v] *= al[v]; acc3[v] *= al[v];
    }

    // P A-fragment from LDS (K dim = 32 local tokens)
    bf16x16 aP;
    {
      bf16x8 lo = *(const bf16x8*)(&s_P[r][16 * hi]);
      bf16x8 hh = *(const bf16x8*)(&s_P[r][16 * hi + 8]);
      aP = concat8(lo, hh);
    }

    // ctx: acc[t] += P(16x32) * V(32x16)
    acc0 = wmma_bf16(aP, bv0, acc0);
    acc1 = wmma_bf16(aP, bv1, acc1);
    acc2 = wmma_bf16(aP, bv2, acc2);
    acc3 = wmma_bf16(aP, bv3, acc3);
    __syncthreads();
  }

  // epilogue: ctx[b][n][c] = acc / l
  float linv[8];
  #pragma unroll
  for (int v = 0; v < 8; ++v) linv[v] = 1.f / s_l[v + 8 * hi];
  float* crow = ctx + ((size_t)b * NN + n0) * CC;
  #pragma unroll
  for (int v = 0; v < 8; ++v) {
    float* cp = crow + (size_t)(v + 8 * hi) * CC + r;
    cp[0]  = acc0[v] * linv[v];
    cp[16] = acc1[v] * linv[v];
    cp[32] = acc2[v] * linv[v];
    cp[48] = acc3[v] * linv[v];
  }
}

// ---------------------------------------------------------------------------
// Kernel 3: out[b][o][n] = x[b][o][n] + sum_c w_add[o][c] * ctx[b][n][c]
// ---------------------------------------------------------------------------
__global__ __launch_bounds__(256) void addproj(
    const float* __restrict__ x, const float* __restrict__ w_add,
    const float* __restrict__ ctx, float* __restrict__ out) {
  __shared__ float s_c[64][65];  // [j][c], +1 pad: conflict-free across j
  __shared__ float s_w[64][64];  // broadcast reads within wave

  const int b  = blockIdx.x / (NN / 64);
  const int n0 = (blockIdx.x % (NN / 64)) * 64;
  const int t  = threadIdx.x;

  for (int i = t; i < 64 * 64; i += 256) s_w[i >> 6][i & 63] = w_add[i];
  for (int i = t; i < 64 * 64; i += 256) {
    const int j = i >> 6, c = i & 63;
    s_c[j][c] = ctx[((size_t)b * NN + n0 + j) * CC + c];
  }
  __syncthreads();

  const int j = t & 63, g = t >> 6;
  for (int oo = 0; oo < 16; ++oo) {
    const int o = g * 16 + oo;
    float a = 0.f;
    #pragma unroll
    for (int c = 0; c < CC; ++c) a = fmaf(s_w[o][c], s_c[j][c], a);
    const size_t idx = ((size_t)b * CC + o) * NN + n0 + j;   // coalesced in j
    out[idx] = x[idx] + a;
  }
}

// ---------------------------------------------------------------------------
extern "C" void kernel_launch(void* const* d_in, const int* in_sizes, int n_in,
                              void* d_out, int out_size, void* d_ws, size_t ws_size,
                              hipStream_t stream) {
  (void)in_sizes; (void)n_in; (void)out_size; (void)ws_size;
  const float* x    = (const float*)d_in[0];
  const float* w1   = (const float*)d_in[1];
  const float* w2   = (const float*)d_in[2];
  const float* w3   = (const float*)d_in[3];
  const float* wadd = (const float*)d_in[4];
  float* out = (float*)d_out;

  const size_t BNC = (size_t)BB * NN * CC;       // 1,605,632 elements
  u16*   Qb  = (u16*)d_ws;                       // bf16 [B][N][C]
  u16*   Kb  = Qb + BNC;                         // bf16 [B][N][C]
  u16*   Vb  = Kb + BNC;                         // bf16 [B][C][N]
  float* ctx = (float*)(Vb + BNC);               // f32  [B][N][C]
  // total ws: 3*BNC*2 + BNC*4 = ~15.3 MiB

  proj_qkv  <<<BB * (NN / 32), 256, 0, stream>>>(x, w1, w2, w3, Qb, Kb, Vb);
  attn_fused<<<BB * (NN / 16),  32, 0, stream>>>(Qb, Kb, Vb, ctx);
  addproj   <<<BB * (NN / 64), 256, 0, stream>>>(x, wadd, ctx, out);
}
